// PRGNN_78469052498048
// MI455X (gfx1250) — compile-verified, matrix operand order checked by reading, not run on hardware
//
#include <hip/hip_runtime.h>
#include <hip/hip_bf16.h>

// ---------------------------------------------------------------------------
// Edge-Conditioned GNN, reformulated as dense bf16 WMMA GEMMs per layer:
//   msg = A_aug (E x 9*Fin) @ W_aug (9*Fin x Fout)     [dominant cost]
//   r   = h (N x Fin) @ Wr (Fin x Fout)
//   h'  = relu(scatter_add(msg, dst) + r + b)
// All K dims are multiples of 32, all M/N dims multiples of 16 -> clean
// v_wmma_f32_16x16x32_bf16 tiling, f32 accumulate.
// GEMM: each wave owns a 64x16 output block (4 M-tiles x 1 N-tile); per
// K-step we batch-load B + all 4 A fragments into distinct registers, then
// fire 4 WMMAs -> loads stay in flight across the math (partial loadcnt
// waits) instead of a full drain before every WMMA.
// ---------------------------------------------------------------------------

typedef __bf16 bf16_t;
typedef __attribute__((ext_vector_type(16))) bf16_t v16bf;
typedef __attribute__((ext_vector_type(8)))  bf16_t v8bf;
typedef __attribute__((ext_vector_type(8)))  float  v8f;

// Load a 16-element A/B fragment for v_wmma_*_16x16x32: per the CDNA5 ISA
// 16-bit A-matrix layout, a lane needs K offsets {0..7} and {16..23} relative
// to its half-selected base -> two contiguous 16-byte loads.
__device__ __forceinline__ v16bf load_frag16(const bf16_t* __restrict__ p) {
    v8bf lo = *(const v8bf*)(p);
    v8bf hi = *(const v8bf*)(p + 16);
    v16bf r;
#pragma unroll
    for (int i = 0; i < 8; ++i) { r[i] = lo[i]; r[i + 8] = hi[i]; }
    return r;
}

#define GEMM_WAVES 4   // waves per block
#define GEMM_MT    4   // 16-row M tiles per wave (64 rows)

// C[M,N] = A[M,K] * BT[N,K]^T  (A,BT bf16 row-major; C f32 row-major)
__global__ __launch_bounds__(128) void gemm_bf16_wmma(
    const bf16_t* __restrict__ A, const bf16_t* __restrict__ BT,
    float* __restrict__ C, int M, int N, int K)
{
    const int wave = threadIdx.x >> 5;
    const int lane = threadIdx.x & 31;
    const int tm0 = (blockIdx.x * GEMM_WAVES + wave) * (16 * GEMM_MT);
    const int tn  = blockIdx.y * 16;
    if (tm0 >= M) return;               // wave-uniform: EXEC stays all-ones

    const int kh = lane >> 4;           // K-half selector
    const int q  = lane & 15;           // A row / B col within tile

    const bf16_t* brow = BT + (size_t)(tn + q) * K + kh * 8;
    const bf16_t* arow[GEMM_MT];
#pragma unroll
    for (int t = 0; t < GEMM_MT; ++t)
        arow[t] = A + (size_t)(tm0 + t * 16 + q) * K + kh * 8;

    v8f acc[GEMM_MT] = {};
    for (int k = 0; k < K; k += 32) {
        // ---- load phase: everything in flight before any math ----
        v16bf b = load_frag16(brow + k);
        v16bf a[GEMM_MT];
#pragma unroll
        for (int t = 0; t < GEMM_MT; ++t) a[t] = load_frag16(arow[t] + k);

        // A is streamed (freshly written each layer, not yet cache-hot):
        // pull the next K-block line in early -> global_prefetch_b8
        __builtin_prefetch(arow[0] + k + 64, 0, 1);

        // ---- math phase: 4 WMMAs, loads for later t still landing ----
#pragma unroll
        for (int t = 0; t < GEMM_MT; ++t)
            acc[t] = __builtin_amdgcn_wmma_f32_16x16x32_bf16(
                false, a[t], false, b, (short)0, acc[t], false, false);
    }

    // C/D layout: VGPR r -> row (r + 8*kh), col q
#pragma unroll
    for (int t = 0; t < GEMM_MT; ++t) {
        float* crow = C + (size_t)(tm0 + t * 16 + kh * 8) * N + (tn + q);
#pragma unroll
        for (int r = 0; r < 8; ++r) crow[(size_t)r * N] = acc[t][r];
    }
}

// h0[n,f] = x[n,f] for f < 32 (drop last column of the 33-wide input)
__global__ void strip_x(const float* __restrict__ x, float* __restrict__ h, int N)
{
    int idx = blockIdx.x * blockDim.x + threadIdx.x;
    if (idx >= N * 32) return;
    int n = idx >> 5, f = idx & 31;
    h[idx] = x[n * 33 + f];
}

// W_aug^T (Fout x 9*Fin) bf16: rows of B^T hold K contiguously so the GEMM's
// B fragment loads are the same two 16B loads as A.
__global__ void build_waugT(const float* __restrict__ Wk, const float* __restrict__ bk,
                            bf16_t* __restrict__ WT, int Fin, int Fout)
{
    int K = 9 * Fin;
    int idx = blockIdx.x * blockDim.x + threadIdx.x;
    if (idx >= Fout * K) return;
    int o = idx / K;
    int r = idx - o * K;
    int s = r / Fin;
    int f = r - s * Fin;
    float v = (s < 8) ? Wk[(size_t)s * Fin * Fout + (size_t)f * Fout + o]
                      : bk[(size_t)f * Fout + o];
    WT[idx] = (bf16_t)v;
}

// Wr^T (Fout x Fin) bf16
__global__ void build_wrT(const float* __restrict__ Wr, bf16_t* __restrict__ WT,
                          int Fin, int Fout)
{
    int idx = blockIdx.x * blockDim.x + threadIdx.x;
    if (idx >= Fin * Fout) return;
    int o = idx / Fin, f = idx - o * Fin;
    WT[idx] = (bf16_t)Wr[(size_t)f * Fout + o];
}

// bf16 copy of node features for the Wr GEMM
__global__ void build_hb(const float* __restrict__ h, bf16_t* __restrict__ hb, int n)
{
    int idx = blockIdx.x * blockDim.x + threadIdx.x;
    if (idx < n) hb[idx] = (bf16_t)h[idx];
}

// A_aug (E x 9*Fin) bf16: gather h[src[e]] once, emit 8 edge-scaled copies + 1x
__global__ void build_aaug(const float* __restrict__ h, const float* __restrict__ ef,
                           const int* __restrict__ src, bf16_t* __restrict__ A,
                           int E, int Fin)
{
    int idx = blockIdx.x * blockDim.x + threadIdx.x;   // e*Fin + f
    if (idx >= E * Fin) return;
    int e = idx / Fin, f = idx - e * Fin;
    float hv = h[(size_t)src[e] * Fin + f];
    const float* er = ef + (size_t)e * 8;
    bf16_t* row = A + (size_t)e * 9 * Fin + f;
#pragma unroll
    for (int s = 0; s < 8; ++s) row[s * Fin] = (bf16_t)(er[s] * hv);
    row[8 * Fin] = (bf16_t)hv;
}

// acc[dst[e], o] += msg[e, o]
__global__ void scatter_msg(const float* __restrict__ msg, const int* __restrict__ dst,
                            float* __restrict__ acc, int E, int Fout)
{
    int idx = blockIdx.x * blockDim.x + threadIdx.x;
    if (idx >= E * Fout) return;
    int e = idx / Fout, o = idx - e * Fout;
    atomicAdd(&acc[(size_t)dst[e] * Fout + o], msg[idx]);
}

// h'[n,o] = relu(acc[n,o] + b[o])
__global__ void relu_bias(const float* __restrict__ acc, const float* __restrict__ b,
                          float* __restrict__ h, int N, int Fout)
{
    int idx = blockIdx.x * blockDim.x + threadIdx.x;
    if (idx >= N * Fout) return;
    int o = idx % Fout;
    h[idx] = fmaxf(acc[idx] + b[o], 0.0f);
}

// out[p] = relu(h[ib[p]]@Wd + bd) - relu(h[ia[p]]@Wd + bd), F = 16
__global__ void final_readout(const float* __restrict__ h, const float* __restrict__ Wd,
                              const float* __restrict__ bd, const int* __restrict__ ia,
                              const int* __restrict__ ib, float* __restrict__ out, int P)
{
    int p = blockIdx.x * blockDim.x + threadIdx.x;
    if (p >= P) return;
    const float* ha = h + (size_t)ia[p] * 16;
    const float* hb = h + (size_t)ib[p] * 16;
    float sa = bd[0], sb = bd[0];
#pragma unroll
    for (int f = 0; f < 16; ++f) { sa += ha[f] * Wd[f]; sb += hb[f] * Wd[f]; }
    out[p] = fmaxf(sb, 0.0f) - fmaxf(sa, 0.0f);
}

static inline int cdiv(int a, int b) { return (a + b - 1) / b; }

extern "C" void kernel_launch(void* const* d_in, const int* in_sizes, int n_in,
                              void* d_out, int out_size, void* d_ws, size_t ws_size,
                              hipStream_t stream)
{
    (void)in_sizes; (void)n_in; (void)out_size; (void)ws_size;
    const int N = 2048, E = 8192, P = 4096;
    const int FD[6] = {32, 256, 128, 64, 32, 16};

    const float* x     = (const float*)d_in[0];
    const float* efeat = (const float*)d_in[1];
    const int*   src   = (const int*)d_in[2];
    const int*   dst   = (const int*)d_in[3];
    const int*   idx_a = (const int*)d_in[5];
    const int*   idx_b = (const int*)d_in[6];
    const float* Wd    = (const float*)d_in[27];
    const float* bd    = (const float*)d_in[28];

    // Workspace layout (256B-aligned slots; total ~55 MB)
    uint8_t* ws = (uint8_t*)d_ws;
    size_t off = 0;
    auto take = [&](size_t bytes) -> void* {
        void* p = ws + off;
        off = (off + bytes + 255) & ~(size_t)255;
        return p;
    };
    float*  H0   = (float*)take((size_t)N * 256 * 4);
    float*  H1   = (float*)take((size_t)N * 256 * 4);
    float*  ACC  = (float*)take((size_t)N * 256 * 4);
    float*  MSG  = (float*)take((size_t)E * 256 * 4);
    bf16_t* AAUG = (bf16_t*)take((size_t)E * 2304 * 2);
    bf16_t* WTA  = (bf16_t*)take((size_t)2304 * 256 * 2);
    bf16_t* WRT  = (bf16_t*)take((size_t)256 * 256 * 2);
    bf16_t* HB   = (bf16_t*)take((size_t)N * 256 * 2);

    strip_x<<<cdiv(N * 32, 256), 256, 0, stream>>>(x, H0, N);

    const int ROWS_PER_BLOCK = 16 * GEMM_MT * GEMM_WAVES;   // 256

    float* hc = H0;
    float* hn = H1;
    for (int l = 0; l < 5; ++l) {
        const int Fin = FD[l], Fout = FD[l + 1], K = 9 * Fin;
        const float* Wk = (const float*)d_in[7 + 4 * l + 0];
        const float* bk = (const float*)d_in[7 + 4 * l + 1];
        const float* Wr = (const float*)d_in[7 + 4 * l + 2];
        const float* b  = (const float*)d_in[7 + 4 * l + 3];

        build_waugT<<<cdiv(Fout * K, 256), 256, 0, stream>>>(Wk, bk, WTA, Fin, Fout);
        build_wrT  <<<cdiv(Fin * Fout, 256), 256, 0, stream>>>(Wr, WRT, Fin, Fout);
        build_hb   <<<cdiv(N * Fin, 256), 256, 0, stream>>>(hc, HB, N * Fin);
        build_aaug <<<cdiv(E * Fin, 256), 256, 0, stream>>>(hc, efeat, src, AAUG, E, Fin);

        // msg = A_aug @ W_aug   (E x K) @ (K x Fout)
        {
            dim3 grid(E / ROWS_PER_BLOCK, Fout / 16);
            gemm_bf16_wmma<<<grid, 128, 0, stream>>>(AAUG, WTA, MSG, E, Fout, K);
        }
        // acc = h @ Wr          (N x Fin) @ (Fin x Fout)
        {
            dim3 grid(N / ROWS_PER_BLOCK, Fout / 16);
            gemm_bf16_wmma<<<grid, 128, 0, stream>>>(HB, WRT, ACC, N, Fout, Fin);
        }

        scatter_msg<<<cdiv(E * Fout, 256), 256, 0, stream>>>(MSG, dst, ACC, E, Fout);
        relu_bias  <<<cdiv(N * Fout, 256), 256, 0, stream>>>(ACC, b, hn, N, Fout);

        float* t = hc; hc = hn; hn = t;
    }

    final_readout<<<cdiv(P, 256), 256, 0, stream>>>(hc, Wd, bd, idx_a, idx_b,
                                                    (float*)d_out, P);
}
